// Base_STAttention_86105504350544
// MI455X (gfx1250) — matrix operand order repaired and checked by reading, not compiled, not run on hardware
//
#include <hip/hip_runtime.h>
#include <hip/hip_bf16.h>

typedef __attribute__((ext_vector_type(16))) _Float16 v16h;
typedef __attribute__((ext_vector_type(8)))  _Float16 v8h;
typedef __attribute__((ext_vector_type(4)))  _Float16 v4h;
typedef __attribute__((ext_vector_type(8)))  float    v8f;
typedef __attribute__((ext_vector_type(4)))  float    v4f;

#define BB   16
#define TTT  256
#define VVV  25
#define DDD  256
#define HHH  8
#define HDIM 32
#define NNN  (BB * TTT * VVV)   // 102400 rows

#if __has_builtin(__builtin_amdgcn_sched_barrier)
#define SCHED_SPLIT() __builtin_amdgcn_sched_barrier(0)
#else
#define SCHED_SPLIT() asm volatile("" ::: "memory")
#endif

static __device__ __forceinline__ v8f wmma_f16(v16h a, v16h b, v8f c) {
  // D = A(16x32 f16) * B(32x16 f16) + C(16x16 f32)
  return __builtin_amdgcn_wmma_f32_16x16x32_f16(false, a, false, b, (short)0, c,
                                                false, false);
}

// A fragment: 16(M) x 32(K), from row-major f16 [m][k], row stride rs elems.
// Lane layout (ISA 7.12.2): lane m=l&15; lanes 0-15 hold K {0..7,16..23},
// lanes 16-31 hold K {8..15,24..31} -> two contiguous 8xf16 chunks per lane.
static __device__ __forceinline__ v16h afrag(const _Float16* A, int rs, int lane) {
  int m  = lane & 15;
  int kb = (lane >> 4) << 3;
  union { v16h v; v8h h2[2]; } u;
  u.h2[0] = *(const v8h*)(A + (size_t)m * rs + kb);
  u.h2[1] = *(const v8h*)(A + (size_t)m * rs + 16 + kb);
  return u.v;
}

// B fragment: 32(K) x 16(N), from memory holding B^T row-major: Bt[n][k],
// row stride rs. Lane n=l&15 holds K {kb..kb+7, 16+kb..16+kb+7}, kb=(l>>4)*8.
static __device__ __forceinline__ v16h bfrag(const _Float16* Bt, int rs, int lane) {
  int n  = lane & 15;
  int kb = (lane >> 4) << 3;
  union { v16h v; v8h h2[2]; } u;
  u.h2[0] = *(const v8h*)(Bt + (size_t)n * rs + kb);
  u.h2[1] = *(const v8h*)(Bt + (size_t)n * rs + 16 + kb);
  return u.v;
}

// ---------------- Kernel 1a: convert x (f32 -> f16), 4 elems/thread
__global__ __launch_bounds__(256) void k_cvt_x(const float* __restrict__ x,
                                               _Float16* __restrict__ x16) {
  size_t i = ((size_t)blockIdx.x * 256 + threadIdx.x) * 4;
  v4f a = *(const v4f*)(x + i);
  v4h o;
#pragma unroll
  for (int j = 0; j < 4; ++j) o[j] = (_Float16)a[j];
  *(v4h*)(x16 + i) = o;
}

// ---------------- Kernel 1b: convert 8 weights (f32 -> f16) --------
// grid = (64, 8); blockIdx.y selects weight slot; 4 elems/thread.
__global__ __launch_bounds__(256) void k_cvt_w(
    const float* w0, const float* w1, const float* w2, const float* w3,
    const float* w4, const float* w5, const float* w6, const float* w7,
    _Float16* __restrict__ w16) {
  const float* w;
  switch (blockIdx.y) {
    case 0: w = w0; break; case 1: w = w1; break;
    case 2: w = w2; break; case 3: w = w3; break;
    case 4: w = w4; break; case 5: w = w5; break;
    case 6: w = w6; break; default: w = w7; break;
  }
  int off = (blockIdx.x * 256 + threadIdx.x) * 4;    // 0..65532
  v4f a = *(const v4f*)(w + off);
  v4h o;
#pragma unroll
  for (int j = 0; j < 4; ++j) o[j] = (_Float16)a[j];
  *(v4h*)(w16 + (size_t)blockIdx.y * DDD * DDD + off) = o;
}

// ---------------- Kernel 2: fused 6-way QKV projection -------------
// Block = 16-row strip of x. 8 waves x 12 tiles = 6 proj * 16 col-tiles.
// proj: 0 qt(scaled) 1 kt 2 vt(->vtT) 3 qs(scaled) 4 ks 5 vs(->vsT)
__global__ __launch_bounds__(256) void k_qkv(
    const _Float16* __restrict__ x16, const _Float16* __restrict__ w16,
    const float* __restrict__ bqt, const float* __restrict__ bkt,
    const float* __restrict__ bvt, const float* __restrict__ bqs,
    const float* __restrict__ bks, const float* __restrict__ bvs,
    _Float16* __restrict__ qt, _Float16* __restrict__ kt,
    _Float16* __restrict__ vtT, _Float16* __restrict__ qs,
    _Float16* __restrict__ ks, _Float16* __restrict__ vsT) {
  __shared__ __align__(64) _Float16 ldsA[16 * DDD];
  int strip = blockIdx.x;
  int tid   = threadIdx.x;
  { // stage 16x256 f16 strip: 256 threads x one v16h each
    const v16h* src = (const v16h*)(x16 + (size_t)strip * 16 * DDD);
    ((v16h*)ldsA)[tid] = src[tid];
  }
  __syncthreads();
  int wave = __builtin_amdgcn_readfirstlane(tid >> 5);  // scalar wave id
  int lane = tid & 31;
  int col = lane & 15, half8 = (lane >> 4) << 3;
  const float kScale = 0.17677669529663687f;  // 1/sqrt(32)

  // Preload all 8 A fragments once (stay in VGPRs across the tile loop).
  v16h afr[8];
#pragma unroll
  for (int kk = 0; kk < 8; ++kk) afr[kk] = afrag(ldsA + kk * 32, DDD, lane);

  for (int tt = 0; tt < 12; ++tt) {
    int g = wave * 12 + tt;                 // scalar
    int proj = g >> 4, ct = g & 15;         // scalar
    const _Float16* wp = w16 + (size_t)proj * DDD * DDD + (size_t)ct * 16 * DDD;

    // Batch-load the full B K-stream; scheduling barrier keeps the 16 b128
    // loads grouped (one wait) and the 8 WMMAs as a back-to-back burst.
    v16h bfr[8];
#pragma unroll
    for (int kk = 0; kk < 8; ++kk) bfr[kk] = bfrag(wp + kk * 32, DDD, lane);
    SCHED_SPLIT();
    v8f acc0 = {}, acc1 = {};
#pragma unroll
    for (int kk = 0; kk < 8; kk += 2) {     // two accumulate chains for ILP
      acc0 = wmma_f16(afr[kk],     bfr[kk],     acc0);
      acc1 = wmma_f16(afr[kk + 1], bfr[kk + 1], acc1);
    }
    v8f acc = acc0 + acc1;

    int colg = ct * 16 + col;
    float bias = (proj == 0) ? bqt[colg] : (proj == 1) ? bkt[colg]
               : (proj == 2) ? bvt[colg] : (proj == 3) ? bqs[colg]
               : (proj == 4) ? bks[colg] : bvs[colg];
    if (proj == 0 || proj == 3) {           // scalar branch
      _Float16* dst = (proj == 0) ? qt : qs;
#pragma unroll
      for (int e = 0; e < 8; ++e) {
        int n = strip * 16 + e + half8;
        dst[(size_t)n * DDD + colg] = (_Float16)((acc[e] + bias) * kScale);
      }
    } else if (proj == 1 || proj == 4) {
      _Float16* dst = (proj == 1) ? kt : ks;
#pragma unroll
      for (int e = 0; e < 8; ++e) {
        int n = strip * 16 + e + half8;
        dst[(size_t)n * DDD + colg] = (_Float16)(acc[e] + bias);
      }
    } else if (proj == 2) {                 // vt transposed: [b,v,h,hd,t]
      int h = colg >> 5, hd = colg & 31;
#pragma unroll
      for (int e = 0; e < 8; ++e) {
        int n = strip * 16 + e + half8;
        int b = n / (TTT * VVV); int r = n % (TTT * VVV);
        int t = r / VVV; int v = r % VVV;
        vtT[(((size_t)(b * VVV + v) * HHH + h) * HDIM + hd) * TTT + t] =
            (_Float16)(acc[e] + bias);
      }
    } else {                                // vs transposed: [b*t,h,hd,32pad]
      int h = colg >> 5, hd = colg & 31;
#pragma unroll
      for (int e = 0; e < 8; ++e) {
        int n = strip * 16 + e + half8;
        int bt = n / VVV; int v = n % VVV;
        vsT[(((size_t)bt * HHH + h) * HDIM + hd) * 32 + v] =
            (_Float16)(acc[e] + bias);
      }
    }
  }
}

// ---------------- Kernel 3: temporal causal flash attention --------
// Block per (b,v,h); 4 waves, wave w owns query tiles {w, w+4, w+8, w+12}.
__global__ __launch_bounds__(128) void k_tatt(
    const _Float16* __restrict__ qt, const _Float16* __restrict__ kt,
    const _Float16* __restrict__ vtT, _Float16* __restrict__ at) {
  __shared__ __align__(64) _Float16 ldsP[4 * 16 * 32];
  int h  = blockIdx.x & 7;
  int bv = blockIdx.x >> 3;               // b*25 + v
  int b = bv / VVV, v = bv % VVV;
  int wave = __builtin_amdgcn_readfirstlane(threadIdx.x >> 5);
  int lane = threadIdx.x & 31;
  int col = lane & 15, half8 = (lane >> 4) << 3;

  const int RS = VVV * DDD;               // 6400: row stride per t
  const _Float16* qbase = qt + ((size_t)(b * TTT) * VVV + v) * DDD + h * HDIM;
  const _Float16* kbase = kt + ((size_t)(b * TTT) * VVV + v) * DDD + h * HDIM;
  const _Float16* vbase = vtT + ((size_t)bv * HHH + h) * (size_t)HDIM * TTT;
  _Float16* obase = at + ((size_t)(b * TTT) * VVV + v) * DDD + h * HDIM;
  _Float16* myP = ldsP + wave * (16 * 32);

  for (int qtile = wave; qtile < 16; qtile += 4) {
    v16h aq = afrag(qbase + (size_t)qtile * 16 * RS, RS, lane);
    float mrow[8], lrow[8];
    v8f o0 = {}, o1 = {};
#pragma unroll
    for (int e = 0; e < 8; ++e) { mrow[e] = -3.0e38f; lrow[e] = 0.0f; }

    int nch = (qtile + 2) >> 1;           // 32-key chunks, causal
    for (int kc = 0; kc < nch; ++kc) {
      v16h bk0 = bfrag(kbase + (size_t)(kc * 32) * RS, RS, lane);
      v16h bk1 = bfrag(kbase + (size_t)(kc * 32 + 16) * RS, RS, lane);
      v8f z = {};
      v8f s0 = wmma_f16(aq, bk0, z);
      v8f s1 = wmma_f16(aq, bk1, z);
      int gk0 = kc * 32 + col, gk1 = gk0 + 16;
      float corr[8];
#pragma unroll
      for (int e = 0; e < 8; ++e) {
        int grow = qtile * 16 + e + half8;
        float a0 = (gk0 <= grow) ? s0[e] : -3.0e38f;
        float a1 = (gk1 <= grow) ? s1[e] : -3.0e38f;
        float mx = fmaxf(a0, a1);
#pragma unroll
        for (int mshf = 1; mshf < 16; mshf <<= 1)
          mx = fmaxf(mx, __shfl_xor(mx, mshf, 16));
        float mnew = fmaxf(mrow[e], mx);
        float p0 = __expf(a0 - mnew);
        float p1 = __expf(a1 - mnew);
        float rsum = p0 + p1;
#pragma unroll
        for (int mshf = 1; mshf < 16; mshf <<= 1)
          rsum += __shfl_xor(rsum, mshf, 16);
        corr[e] = __expf(mrow[e] - mnew);
        mrow[e] = mnew;
        lrow[e] = lrow[e] * corr[e] + rsum;
        int row = e + half8;
        myP[row * 32 + col]      = (_Float16)p0;
        myP[row * 32 + 16 + col] = (_Float16)p1;
        o0[e] *= corr[e];
        o1[e] *= corr[e];
      }
      asm volatile("s_wait_dscnt 0" ::: "memory");   // P writes visible to wave
      v16h ap  = afrag(myP, 32, lane);
      v16h bv0 = bfrag(vbase + kc * 32, TTT, lane);              // hd 0..15
      v16h bv1 = bfrag(vbase + (size_t)16 * TTT + kc * 32, TTT, lane); // hd 16..31
      o0 = wmma_f16(ap, bv0, o0);
      o1 = wmma_f16(ap, bv1, o1);
    }
#pragma unroll
    for (int e = 0; e < 8; ++e) {
      float inv = 1.0f / lrow[e];
      int t = qtile * 16 + e + half8;
      _Float16* orow = obase + (size_t)t * RS;
      orow[col]      = (_Float16)(o0[e] * inv);
      orow[16 + col] = (_Float16)(o1[e] * inv);
    }
  }
}

// ---------------- Kernel 4: spatial attention (V=25, padded to 32) -
// Block per (b,t); 8 waves = 8 heads.
__global__ __launch_bounds__(256) void k_satt(
    const _Float16* __restrict__ qs, const _Float16* __restrict__ ks,
    const _Float16* __restrict__ vsT, _Float16* __restrict__ as_) {
  __shared__ __align__(64) _Float16 ldsP[8 * 16 * 32];
  int bt = blockIdx.x;
  int wave = __builtin_amdgcn_readfirstlane(threadIdx.x >> 5);
  int lane = threadIdx.x & 31;
  int h = wave;                           // scalar head id
  int col = lane & 15, half8 = (lane >> 4) << 3;
  const _Float16* qbase = qs + (size_t)bt * VVV * DDD + h * HDIM;
  const _Float16* kbase = ks + (size_t)bt * VVV * DDD + h * HDIM;
  const _Float16* vbase = vsT + ((size_t)bt * HHH + h) * (HDIM * 32);
  _Float16* obase = as_ + (size_t)bt * VVV * DDD + h * HDIM;
  _Float16* myP = ldsP + wave * (16 * 32);

  v16h bk0 = bfrag(kbase, DDD, lane);     // keys 0..15 all valid
  v16h bk1 = {};
  {
    int key = 16 + col;
    if (key < VVV)
      bk1 = bfrag(kbase + (size_t)16 * DDD, DDD, lane);
  }
  v16h bvs0 = bfrag(vbase, 32, lane);                   // hd 0..15 (pad v zeroed)
  v16h bvs1 = bfrag(vbase + (size_t)16 * 32, 32, lane); // hd 16..31

  for (int qi = 0; qi < 2; ++qi) {
    v16h aq = {};
    {
      int gv = qi * 16 + (lane & 15);
      if (gv < VVV) {
        int kb = (lane >> 4) << 3;
        union { v16h vv; v8h h2[2]; } u;
        const _Float16* p = qbase + (size_t)gv * DDD;
        u.h2[0] = *(const v8h*)(p + kb);
        u.h2[1] = *(const v8h*)(p + 16 + kb);
        aq = u.vv;
      }
    }
    v8f z = {};
    v8f s0 = wmma_f16(aq, bk0, z);
    v8f s1 = wmma_f16(aq, bk1, z);
    bool k1ok = (16 + col) < VVV;
#pragma unroll
    for (int e = 0; e < 8; ++e) {
      float a0 = s0[e];
      float a1 = k1ok ? s1[e] : -3.0e38f;
      float mx = fmaxf(a0, a1);
#pragma unroll
      for (int mshf = 1; mshf < 16; mshf <<= 1)
        mx = fmaxf(mx, __shfl_xor(mx, mshf, 16));
      float p0 = __expf(a0 - mx);
      float p1 = __expf(a1 - mx);
      float rsum = p0 + p1;
#pragma unroll
      for (int mshf = 1; mshf < 16; mshf <<= 1)
        rsum += __shfl_xor(rsum, mshf, 16);
      float inv = 1.0f / rsum;
      int row = e + half8;
      myP[row * 32 + col]      = (_Float16)(p0 * inv);
      myP[row * 32 + 16 + col] = (_Float16)(p1 * inv);
    }
    asm volatile("s_wait_dscnt 0" ::: "memory");
    v16h ap = afrag(myP, 32, lane);
    v8f z2 = {};
    v8f o0 = wmma_f16(ap, bvs0, z2);
    v8f o1 = wmma_f16(ap, bvs1, z2);
#pragma unroll
    for (int e = 0; e < 8; ++e) {
      int gv = qi * 16 + e + half8;
      if (gv < VVV) {
        _Float16* orow = obase + (size_t)gv * DDD;
        orow[col]      = (_Float16)o0[e];
        orow[16 + col] = (_Float16)o1[e];
      }
    }
  }
}

// ---------------- Kernel 5: fused output projection + add ----------
// out[n,:] = at[n]@wpt^T + as[n]@wps^T + bpt + bps   (f32 out)
__global__ __launch_bounds__(256) void k_outproj(
    const _Float16* __restrict__ at, const _Float16* __restrict__ as_,
    const _Float16* __restrict__ w16, const float* __restrict__ bpt,
    const float* __restrict__ bps, float* __restrict__ out) {
  __shared__ __align__(64) _Float16 ldsA[2 * 16 * DDD];
  int strip = blockIdx.x, tid = threadIdx.x;
  {
    ((v16h*)ldsA)[tid]       = ((const v16h*)(at  + (size_t)strip * 16 * DDD))[tid];
    ((v16h*)ldsA)[256 + tid] = ((const v16h*)(as_ + (size_t)strip * 16 * DDD))[tid];
  }
  __syncthreads();
  int wave = __builtin_amdgcn_readfirstlane(tid >> 5);
  int lane = tid & 31;
  int col = lane & 15, half8 = (lane >> 4) << 3;
  const _Float16* wpt16 = w16 + (size_t)6 * DDD * DDD;
  const _Float16* wps16 = w16 + (size_t)7 * DDD * DDD;

  // Hoist both A-fragment sets out of the tile loop (LDS read once).
  v16h afrT[8], afrS[8];
#pragma unroll
  for (int kk = 0; kk < 8; ++kk) {
    afrT[kk] = afrag(ldsA + kk * 32, DDD, lane);
    afrS[kk] = afrag(ldsA + 16 * DDD + kk * 32, DDD, lane);
  }

  for (int tt = 0; tt < 2; ++tt) {
    int ct = wave * 2 + tt;
    v16h bfr[8];
    v8f acc0 = {}, acc1 = {};
#pragma unroll
    for (int kk = 0; kk < 8; ++kk)
      bfr[kk] = bfrag(wpt16 + (size_t)ct * 16 * DDD + kk * 32, DDD, lane);
    SCHED_SPLIT();
#pragma unroll
    for (int kk = 0; kk < 8; kk += 2) {
      acc0 = wmma_f16(afrT[kk],     bfr[kk],     acc0);
      acc1 = wmma_f16(afrT[kk + 1], bfr[kk + 1], acc1);
    }
#pragma unroll
    for (int kk = 0; kk < 8; ++kk)
      bfr[kk] = bfrag(wps16 + (size_t)ct * 16 * DDD + kk * 32, DDD, lane);
    SCHED_SPLIT();
#pragma unroll
    for (int kk = 0; kk < 8; kk += 2) {
      acc0 = wmma_f16(afrS[kk],     bfr[kk],     acc0);
      acc1 = wmma_f16(afrS[kk + 1], bfr[kk + 1], acc1);
    }
    v8f acc = acc0 + acc1;

    int colg = ct * 16 + col;
    float bias = bpt[colg] + bps[colg];
#pragma unroll
    for (int e = 0; e < 8; ++e) {
      int n = strip * 16 + e + half8;
      out[(size_t)n * DDD + colg] = acc[e] + bias;
    }
  }
}

// ---------------- host launcher ------------------------------------
extern "C" void kernel_launch(void* const* d_in, const int* in_sizes, int n_in,
                              void* d_out, int out_size, void* d_ws, size_t ws_size,
                              hipStream_t stream) {
  const float* x   = (const float*)d_in[0];
  const float* wkt = (const float*)d_in[1];  const float* bkt = (const float*)d_in[2];
  const float* wqt = (const float*)d_in[3];  const float* bqt = (const float*)d_in[4];
  const float* wvt = (const float*)d_in[5];  const float* bvt = (const float*)d_in[6];
  const float* wpt = (const float*)d_in[7];  const float* bpt = (const float*)d_in[8];
  const float* wks = (const float*)d_in[9];  const float* bks = (const float*)d_in[10];
  const float* wqs = (const float*)d_in[11]; const float* bqs = (const float*)d_in[12];
  const float* wvs = (const float*)d_in[13]; const float* bvs = (const float*)d_in[14];
  const float* wps = (const float*)d_in[15]; const float* bps = (const float*)d_in[16];

  const size_t ND = (size_t)NNN * DDD;               // 26,214,400
  _Float16* ws  = (_Float16*)d_ws;
  _Float16* x16 = ws;                                // dead after k_qkv -> reused as at
  _Float16* w16 = ws + ND;                           // 8 * 64K = 524,288
  _Float16* qtb = w16 + (size_t)8 * DDD * DDD;       // dead after k_tatt -> reused as as
  _Float16* ktb = qtb + ND;
  _Float16* qsb = ktb + ND;
  _Float16* ksb = qsb + ND;
  _Float16* vtT = ksb + ND;                          // [b,v,h,hd,t]
  _Float16* vsT = vtT + ND;                          // [b*t,h,hd,32]  33,554,432
  _Float16* at  = x16;
  _Float16* as_ = qtb;
  const size_t vsT_bytes = (size_t)BB * TTT * HHH * HDIM * 32 * sizeof(_Float16);

  // Zero vsT so its v=25..31 pad can never inject NaN into the P*V WMMA
  // (0 x NaN = NaN would poison entire output tiles on the first call).
  hipMemsetAsync((void*)vsT, 0, vsT_bytes, stream);

  k_cvt_x<<<(unsigned)(ND / (256 * 4)), 256, 0, stream>>>(x, x16);
  k_cvt_w<<<dim3(64, 8), 256, 0, stream>>>(wqt, wkt, wvt, wqs, wks, wvs, wpt, wps, w16);
  k_qkv<<<NNN / 16, 256, 0, stream>>>(x16, w16, bqt, bkt, bvt, bqs, bks, bvs,
                                      qtb, ktb, vtT, qsb, ksb, vsT);
  k_tatt<<<BB * VVV * HHH, 128, 0, stream>>>(qtb, ktb, vtT, at);
  k_satt<<<BB * TTT, 256, 0, stream>>>(qsb, ksb, vsT, as_);
  k_outproj<<<NNN / 16, 256, 0, stream>>>(at, as_, w16, bpt, bps, (float*)d_out);
}
